// Attention_2456721293969
// MI455X (gfx1250) — matrix operand order repaired and checked by reading, not compiled, not run on hardware
//
#include <hip/hip_runtime.h>
#include <hip/hip_bf16.h>
#include <math.h>

#define B_  2
#define T_  2048
#define HID 2048
#define NH  32
#define NKV 8
#define HD  64

typedef __bf16 bf16_t;
typedef bf16_t v8bf  __attribute__((ext_vector_type(8)));
typedef bf16_t v16bf __attribute__((ext_vector_type(16)));
typedef float  v8f   __attribute__((ext_vector_type(8)));

// ---------------------------------------------------------------------------
// WMMA helpers (CDNA5 wave32, D = A(16x32 bf16) * B(32x16 bf16) + C(16x16 f32))
// ---------------------------------------------------------------------------
static __device__ inline v8f wmma_bf16(v16bf a, v16bf b, v8f c) {
  // (neg_a, A, neg_b, B, c_mod, C, reuse_a, reuse_b)
  return __builtin_amdgcn_wmma_f32_16x16x32_bf16(false, a, false, b, (short)0, c,
                                                 false, false);
}

// A fragment, 16x32 bf16, row-major source with leading dim lda.
// Contiguous k order (same as load_bT_frag): lane = row (lane&15); lanes 0-15 hold K = k0+0..15,
// lanes 16-31 hold K = k0+16..31 (two contiguous 16B loads).
static __device__ inline v16bf load_a_frag(const bf16_t* A, int lda, int k0) {
  const int lane = threadIdx.x & 31;
  const int m    = lane & 15;
  const int half = lane >> 4;
  const bf16_t* p = A + (size_t)m * lda + k0 + half * 16;
  v8bf lo = *(const v8bf*)p;
  v8bf hi = *(const v8bf*)(p + 8);
  v16bf f;
#pragma unroll
  for (int i = 0; i < 8; ++i) { f[i] = lo[i]; f[i + 8] = hi[i]; }
  return f;
}

// B fragment for C = A * W^T where W is row-major (N x K, leading dim ldw):
// B[kk][n] = W[n0+n][k0+kk].  Contiguous k order (same as load_a_frag): lane = column n (lane&15);
// lanes 0-15 hold K = k0+0..15, lanes 16-31 hold K = k0+16..31 -> one 32B load.
static __device__ inline v16bf load_bT_frag(const bf16_t* W, int ldw, int n0, int k0) {
  const int lane = threadIdx.x & 31;
  const int n    = n0 + (lane & 15);
  const int half = lane >> 4;
  return *(const v16bf*)(W + (size_t)n * ldw + k0 + half * 16);
}

// ---------------------------------------------------------------------------
// f32 -> bf16 elementwise conversion
// ---------------------------------------------------------------------------
__global__ void cvt_f32_bf16(const float* __restrict__ s, bf16_t* __restrict__ d, int n) {
  int i = blockIdx.x * blockDim.x + threadIdx.x;
  if (i < n) d[i] = (bf16_t)s[i];
}

// ---------------------------------------------------------------------------
// GEMM: C(MxN, f32) = A(MxK, bf16) * W^T, W row-major (N x K, bf16).
// One wave computes a 32x64 tile of C: 2 A-fragments x 4 B-fragments,
// each B-fragment reused for both row-halves -> 8 WMMA per 6KB of loads
// (~21 FLOP/byte from cache) vs 4 WMMA per 5KB for a 16x64 tile.
// ---------------------------------------------------------------------------
__global__ __launch_bounds__(128) void gemm_bf16_nt(const bf16_t* __restrict__ A,
                                                    const bf16_t* __restrict__ W,
                                                    float* __restrict__ C,
                                                    int M, int N, int K) {
  const int wave   = blockIdx.x * (blockDim.x >> 5) + (threadIdx.x >> 5);
  const int tilesN = N >> 6;
  const int total  = (M >> 5) * tilesN;
  if (wave >= total) return;  // wave-uniform exit: EXEC stays all-ones for WMMA
  const int mt = wave / tilesN;
  const int nt = wave % tilesN;
  const bf16_t* Arow0 = A + (size_t)(mt * 32)      * K;
  const bf16_t* Arow1 = A + (size_t)(mt * 32 + 16) * K;

  v8f acc[2][4];
#pragma unroll
  for (int i = 0; i < 2; ++i)
#pragma unroll
    for (int j = 0; j < 4; ++j) acc[i][j] = (v8f){};

  for (int k0 = 0; k0 < K; k0 += 32) {
    const v16bf a0 = load_a_frag(Arow0, K, k0);
    const v16bf a1 = load_a_frag(Arow1, K, k0);
#pragma unroll
    for (int j = 0; j < 4; ++j) {
      const v16bf b = load_bT_frag(W, K, nt * 64 + j * 16, k0);
      acc[0][j] = wmma_bf16(a0, b, acc[0][j]);
      acc[1][j] = wmma_bf16(a1, b, acc[1][j]);
    }
  }

  const int lane = threadIdx.x & 31;
  const int ln   = lane & 15;
  const int half = lane >> 4;
#pragma unroll
  for (int i = 0; i < 2; ++i) {
#pragma unroll
    for (int r = 0; r < 8; ++r) {
      const int m = mt * 32 + i * 16 + r + half * 8;  // C layout: VGPR r <-> M=r+8*half
      float* row = C + (size_t)m * N + nt * 64;
      row[ln]      = acc[i][0][r];
      row[16 + ln] = acc[i][1][r];
      row[32 + ln] = acc[i][2][r];
      row[48 + ln] = acc[i][3][r];
    }
  }
}

// ---------------------------------------------------------------------------
// RoPE + relayout: src f32 [B*T][nh*64] -> dst bf16 [b][h][t][64], *scale
// ---------------------------------------------------------------------------
__global__ void rope_relayout(const float* __restrict__ src, const float* __restrict__ fc,
                              bf16_t* __restrict__ dst, int nh, float scale) {
  int i = blockIdx.x * blockDim.x + threadIdx.x;
  const int tot = B_ * T_ * 32 * nh;
  if (i >= tot) return;
  const int pr = i & 31;
  const int h  = (i >> 5) % nh;
  const int t  = (i / (32 * nh)) % T_;
  const int b  = i / (32 * nh * T_);
  const float* row = src + (size_t)(b * T_ + t) * (nh * HD) + h * HD;
  const float xr = row[2 * pr];
  const float xi = row[2 * pr + 1];
  const float c = fc[(t * 32 + pr) * 2 + 0];
  const float s = fc[(t * 32 + pr) * 2 + 1];
  bf16_t* d = dst + ((size_t)(b * nh + h) * T_ + t) * HD;
  d[2 * pr]     = (bf16_t)((xr * c - xi * s) * scale);
  d[2 * pr + 1] = (bf16_t)((xr * s + xi * c) * scale);
}

// ---------------------------------------------------------------------------
// V relayout+transpose: src f32 [B*T][NKV*64] -> dst bf16 [b][kvh][d][t]
// (so PV B-fragments become contiguous row loads)
// ---------------------------------------------------------------------------
__global__ void v_transpose(const float* __restrict__ vf, bf16_t* __restrict__ vt) {
  int i = blockIdx.x * blockDim.x + threadIdx.x;
  const int total = B_ * T_ * NKV * HD;
  if (i >= total) return;
  const int col = i % (NKV * HD);
  const int m   = i / (NKV * HD);
  const int b = m / T_, t = m % T_;
  const int kvh = col / HD, d = col % HD;
  vt[((size_t)(b * NKV + kvh) * HD + d) * T_ + t] = (bf16_t)vf[i];
}

// ---------------------------------------------------------------------------
// Flash attention: one wave (block of 32) per (b, head, 16-row query tile).
// Online softmax over 32-key chunks; QK^T and PV both via WMMA bf16.
// ---------------------------------------------------------------------------
__global__ __launch_bounds__(32) void flash_attn(const bf16_t* __restrict__ Qb,
                                                 const bf16_t* __restrict__ Kb,
                                                 const bf16_t* __restrict__ Vt,
                                                 bf16_t* __restrict__ Ob) {
  const int QT  = T_ / 16;
  const int bid = blockIdx.x;
  const int qt  = bid % QT;
  const int h   = (bid / QT) % NH;
  const int b   = bid / (QT * NH);
  const int kvh = h >> 2;                 // n_rep = 4
  const int lane = threadIdx.x & 31;
  const int ln   = lane & 15;
  const int half = lane >> 4;

  const bf16_t* Q  = Qb + ((size_t)(b * NH + h) * T_ + qt * 16) * HD;
  const bf16_t* Kh = Kb + (size_t)(b * NKV + kvh) * T_ * HD;     // [t][d]
  const bf16_t* Vh = Vt + (size_t)(b * NKV + kvh) * HD * T_;     // [d][t]

  const v16bf aq0 = load_a_frag(Q, HD, 0);    // Q already scaled by 1/sqrt(64)
  const v16bf aq1 = load_a_frag(Q, HD, 32);

  v8f o0 = {}, o1 = {}, o2 = {}, o3 = {};
  float mrow[8], lrow[8];
#pragma unroll
  for (int r = 0; r < 8; ++r) { mrow[r] = -1e30f; lrow[r] = 0.0f; }

  __shared__ __align__(32) bf16_t pl[16 * 32];  // P tile staging (C-layout -> A-layout)

  const int nch = (qt * 16 + 15) / 32 + 1;
  for (int c = 0; c < nch; ++c) {
    const int s0 = c * 32;

    v8f S0 = {}, S1 = {};
    S0 = wmma_bf16(aq0, load_bT_frag(Kh, HD, s0,       0), S0);
    S0 = wmma_bf16(aq1, load_bT_frag(Kh, HD, s0,      32), S0);
    S1 = wmma_bf16(aq0, load_bT_frag(Kh, HD, s0 + 16,  0), S1);
    S1 = wmma_bf16(aq1, load_bT_frag(Kh, HD, s0 + 16, 32), S1);

    float alpha[8];
#pragma unroll
    for (int r = 0; r < 8; ++r) {
      const int q = qt * 16 + r + half * 8;
      float x0 = (s0 + ln      > q) ? -1e30f : S0[r];   // causal mask
      float x1 = (s0 + 16 + ln > q) ? -1e30f : S1[r];
      float mx = fmaxf(x0, x1);
      mx = fmaxf(mx, __shfl_xor(mx, 1));   // masks 1/2/4/8 stay within 16-lane half
      mx = fmaxf(mx, __shfl_xor(mx, 2));
      mx = fmaxf(mx, __shfl_xor(mx, 4));
      mx = fmaxf(mx, __shfl_xor(mx, 8));
      const float mnew = fmaxf(mrow[r], mx);
      const float al   = __expf(mrow[r] - mnew);
      const float p0   = __expf(x0 - mnew);
      const float p1   = __expf(x1 - mnew);
      float rs = p0 + p1;
      rs += __shfl_xor(rs, 1);
      rs += __shfl_xor(rs, 2);
      rs += __shfl_xor(rs, 4);
      rs += __shfl_xor(rs, 8);
      lrow[r]  = lrow[r] * al + rs;
      mrow[r]  = mnew;
      alpha[r] = al;
      const int m = r + half * 8;
      pl[m * 32 + ln]      = (bf16_t)p0;
      pl[m * 32 + 16 + ln] = (bf16_t)p1;
    }
#pragma unroll
    for (int r = 0; r < 8; ++r) {
      o0[r] *= alpha[r]; o1[r] *= alpha[r]; o2[r] *= alpha[r]; o3[r] *= alpha[r];
    }
    __syncthreads();                         // single-wave WG: near-free, orders LDS
    const v16bf ap = load_a_frag(pl, 32, 0); // P as A-fragment (16x32)
    o0 = wmma_bf16(ap, load_bT_frag(Vh, T_,  0, s0), o0);
    o1 = wmma_bf16(ap, load_bT_frag(Vh, T_, 16, s0), o1);
    o2 = wmma_bf16(ap, load_bT_frag(Vh, T_, 32, s0), o2);
    o3 = wmma_bf16(ap, load_bT_frag(Vh, T_, 48, s0), o3);
    __syncthreads();
  }

#pragma unroll
  for (int r = 0; r < 8; ++r) {
    const float inv = 1.0f / lrow[r];
    const int m = r + half * 8;
    bf16_t* row = Ob + ((size_t)(b * T_) + qt * 16 + m) * (NH * HD) + h * HD;
    row[ln]      = (bf16_t)(o0[r] * inv);
    row[16 + ln] = (bf16_t)(o1[r] * inv);
    row[32 + ln] = (bf16_t)(o2[r] * inv);
    row[48 + ln] = (bf16_t)(o3[r] * inv);
  }
}

// ---------------------------------------------------------------------------
// Host-side orchestration
// ---------------------------------------------------------------------------
extern "C" void kernel_launch(void* const* d_in, const int* in_sizes, int n_in,
                              void* d_out, int out_size, void* d_ws, size_t ws_size,
                              hipStream_t stream) {
  const float* x  = (const float*)d_in[0];
  const float* fc = (const float*)d_in[1];
  const float* wq = (const float*)d_in[2];
  const float* wk = (const float*)d_in[3];
  const float* wv = (const float*)d_in[4];
  const float* wo = (const float*)d_in[5];
  float* out = (float*)d_out;

  char* ws = (char*)d_ws;
  size_t off = 0;
  auto take = [&](size_t bytes) -> char* {
    char* p = ws + off;
    off += (bytes + 255) & ~(size_t)255;
    return p;
  };
  const int M = B_ * T_;  // 4096

  bf16_t* xb  = (bf16_t*)take((size_t)M * HID * 2);
  bf16_t* wqb = (bf16_t*)take((size_t)(NH  * HD) * HID * 2);
  bf16_t* wkb = (bf16_t*)take((size_t)(NKV * HD) * HID * 2);
  bf16_t* wvb = (bf16_t*)take((size_t)(NKV * HD) * HID * 2);
  bf16_t* wob = (bf16_t*)take((size_t)HID * (NH * HD) * 2);
  float*  qf  = (float*) take((size_t)M * NH  * HD * 4);
  float*  kf  = (float*) take((size_t)M * NKV * HD * 4);
  float*  vf  = (float*) take((size_t)M * NKV * HD * 4);
  bf16_t* qb  = (bf16_t*)take((size_t)M * NH  * HD * 2);
  bf16_t* kb  = (bf16_t*)take((size_t)M * NKV * HD * 2);
  bf16_t* vt  = (bf16_t*)take((size_t)M * NKV * HD * 2);
  bf16_t* ab  = (bf16_t*)take((size_t)M * NH  * HD * 2);

  auto cvt = [&](const float* s, bf16_t* d, int n) {
    cvt_f32_bf16<<<(n + 255) / 256, 256, 0, stream>>>(s, d, n);
  };
  cvt(x,  xb,  M * HID);
  cvt(wq, wqb, NH  * HD * HID);
  cvt(wk, wkb, NKV * HD * HID);
  cvt(wv, wvb, NKV * HD * HID);
  cvt(wo, wob, HID * NH * HD);

  auto gemm = [&](const bf16_t* A, const bf16_t* W, float* C, int Mg, int Ng, int Kg) {
    int waves = (Mg / 32) * (Ng / 64);
    gemm_bf16_nt<<<(waves + 3) / 4, 128, 0, stream>>>(A, W, C, Mg, Ng, Kg);
  };
  gemm(xb, wqb, qf, M, NH  * HD, HID);
  gemm(xb, wkb, kf, M, NKV * HD, HID);
  gemm(xb, wvb, vf, M, NKV * HD, HID);

  {
    int n = M * NH * 32;   // q: fold softmax scale 1/sqrt(64)
    rope_relayout<<<(n + 255) / 256, 256, 0, stream>>>(qf, fc, qb, NH, 0.125f);
    n = M * NKV * 32;
    rope_relayout<<<(n + 255) / 256, 256, 0, stream>>>(kf, fc, kb, NKV, 1.0f);
    n = M * NKV * HD;
    v_transpose<<<(n + 255) / 256, 256, 0, stream>>>(vf, vt);
  }

  flash_attn<<<B_ * NH * (T_ / 16), 32, 0, stream>>>(qb, kb, vt, ab);

  gemm(ab, wob, out, M, NH * HD, HID);

  (void)in_sizes; (void)n_in; (void)out_size; (void)ws_size;
}
